// CoPE_34806414967363
// MI455X (gfx1250) — matrix operand order repaired and checked by reading, not compiled, not run on hardware
//
#include <hip/hip_runtime.h>
#include <hip/hip_bf16.h>

typedef __attribute__((ext_vector_type(16))) __bf16 v16bf;
typedef __attribute__((ext_vector_type(8)))  float  v8f;

#define B_DIM 8
#define T_DIM 2048
#define C_DIM 512
#define TMAX  2048
#define NTILES (TMAX / 16)   // 128 N-tiles of width 16

__device__ __forceinline__ __bf16 f2bf(float f) {
    union { float f; unsigned u; } v; v.f = f;
    // round-to-nearest-even bf16
    unsigned r = (v.u + 0x7FFFu + ((v.u >> 16) & 1u)) >> 16;
    union { unsigned short s; __bf16 b; } o; o.s = (unsigned short)r;
    return o.b;
}

// Kernel 0: pos_emb (C x TMAX, f32 row-major) -> posT (TMAX x C, bf16)
// so a WMMA B-fragment (16 consecutive channels at fixed position n) is one
// contiguous 32-byte load straight out of L2.
__global__ void cope_cvt_posemb(const float* __restrict__ pe,
                                __bf16* __restrict__ posT) {
    int idx = blockIdx.x * 256 + threadIdx.x;   // idx = n*C + c
    int n = idx >> 9;
    int c = idx & (C_DIM - 1);
    posT[idx] = f2bf(pe[(size_t)c * TMAX + n]);
}

// Main fused kernel: one block = 16 rows of one batch.
// LDS: [0, 128KB)      f32 logits tile 16 x 2048
//      [128KB, +16KB)  bf16 Q tile pre-swizzled into A-fragment order
//      [+1KB)          f32 scan buffer (256)
__global__ __launch_bounds__(256)
void cope_main(const float* __restrict__ Q,
               const float* __restrict__ A,
               const __bf16* __restrict__ posT,
               float* __restrict__ out) {
    extern __shared__ char smem[];
    float*  logits = (float*)smem;                                // 16*2048*4
    __bf16* qsw    = (__bf16*)(smem + 16 * TMAX * 4);             // 8192 elems
    float*  scan   = (float*)(smem + 16 * TMAX * 4 + 8192 * 2);   // 256 elems

    const int tid = threadIdx.x;
    const int b   = blockIdx.x >> 7;          // 128 row-tiles per batch
    const int t0  = (blockIdx.x & 127) << 4;

    // ---- Stage Q tile as pre-swizzled bf16 A-fragments ----------------------
    // qsw element layout: [((m*16 + ks)*2 + half)*16 + ele], matching the B loads'
// contiguous K order: VGPR i holds K = half*16 + 2i (+piece);
    // channel = ks*32 + K.
    for (int idx = tid; idx < 16 * 16 * 2 * 16; idx += 256) {
        int ele  = idx & 15;
        int half = (idx >> 4) & 1;
        int ks   = (idx >> 5) & 15;
        int m    = idx >> 9;
        int i = ele >> 1, piece = ele & 1;
        int kb = half * 16 + 2 * i;
        int c  = ks * 32 + kb + piece;
        qsw[idx] = f2bf(Q[((size_t)(b * T_DIM + t0 + m)) * C_DIM + c]);
    }
    __syncthreads();

    // ---- GEMM: logits(16 x 2048) = Qtile(16x512) x pos_emb(512x2048) --------
    const int lane = tid & 31;
    const int wid  = tid >> 5;
    const int nloc = lane & 15;   // N (and A-row m) within tile
    const int half = lane >> 4;
    const v16bf* aw = (const v16bf*)qsw;

    for (int nt = wid; nt < NTILES; nt += 8) {
        v8f acc = {};
        const int n = nt * 16 + nloc;
        // B-fragment: element j = posT[n][ks*32 + half*16 + j] (contiguous)
        const __bf16* bp = posT + (size_t)n * C_DIM + half * 16;
        #pragma unroll
        for (int ks = 0; ks < 16; ++ks) {
            v16bf a  = aw[(nloc * 16 + ks) * 2 + half];
            v16bf bb = *(const v16bf*)(bp + ks * 32);
            acc = __builtin_amdgcn_wmma_f32_16x16x32_bf16(
                false, a, false, bb, (short)0, acc, false, false);
        }
        // C/D layout: VGPR e -> row (half*8 + e), col = n
        #pragma unroll
        for (int e = 0; e < 8; ++e)
            logits[(half * 8 + e) * TMAX + n] = acc[e];
    }
    __syncthreads();

    // ---- Per row: sigmoid -> reverse cumsum -> gather + lerp ----------------
    const int kbase = tid * 8;    // 256 threads x 8 elems = 2048
    for (int r = 0; r < 16; ++r) {
        const float* Arow = A + ((size_t)(b * T_DIM + t0 + r)) * T_DIM;
        float g[8];
        #pragma unroll
        for (int e = 0; e < 8; ++e) {
            float x = Arow[kbase + e];
            g[e] = 1.0f / (1.0f + __expf(-x));
        }
        // within-thread suffix sums (reverse cumsum piece)
        float suf[8]; float run = 0.0f;
        #pragma unroll
        for (int e = 7; e >= 0; --e) { run += g[e]; suf[e] = run; }

        scan[tid] = run;
        __syncthreads();
        // Hillis-Steele inclusive suffix scan over the 256 partials
        for (int off = 1; off < 256; off <<= 1) {
            float add = (tid + off < 256) ? scan[tid + off] : 0.0f;
            __syncthreads();
            scan[tid] += add;
            __syncthreads();
        }
        const float beyond = scan[tid] - run;  // sum over threads > tid

        float* lrow = logits + r * TMAX;
        float* orow = out + ((size_t)(b * T_DIM + t0 + r)) * T_DIM;
        #pragma unroll
        for (int e = 0; e < 8; ++e) {
            float pos = fminf(beyond + suf[e], (float)(TMAX - 1));
            float pf  = floorf(pos);
            int   pfi = (int)pf;
            int   pci = (int)ceilf(pos);
            float w   = pos - pf;
            float lc  = lrow[pci];
            float lf  = lrow[pfi];
            orow[kbase + e] = lc * w + lf * (1.0f - w);
        }
        __syncthreads();
    }
}

extern "C" void kernel_launch(void* const* d_in, const int* in_sizes, int n_in,
                              void* d_out, int out_size, void* d_ws, size_t ws_size,
                              hipStream_t stream) {
    const float* Q  = (const float*)d_in[0];   // (B,T,C)   f32
    const float* A  = (const float*)d_in[1];   // (B,T,T)   f32
    const float* pe = (const float*)d_in[2];   // (1,C,TMAX) f32
    float* out = (float*)d_out;                // (B,T,T)   f32
    __bf16* posT = (__bf16*)d_ws;              // 2 MB bf16 transposed pos_emb

    const size_t shmem = (size_t)16 * TMAX * 4 + 8192 * 2 + 256 * 4; // ~148 KB
    // Opt in to >64KB dynamic LDS (CDNA5 allows up to 320 KB per workgroup).
    (void)hipFuncSetAttribute((const void*)cope_main,
                              hipFuncAttributeMaxDynamicSharedMemorySize,
                              (int)shmem);

    cope_cvt_posemb<<<(C_DIM * TMAX) / 256, 256, 0, stream>>>(pe, posT);
    cope_main<<<B_DIM * (T_DIM / 16), 256, shmem, stream>>>(Q, A, posT, out);
}
